// BilingualSentenceClassifier_21818433864103
// MI455X (gfx1250) — matrix operand order repaired and checked by reading, not compiled, not run on hardware
//
#include <hip/hip_runtime.h>
#include <hip/hip_fp16.h>
#include <math.h>

// ---------------------------------------------------------------------------
// BilingualSentenceClassifier for MI455X (gfx1250, wave32).
// Memory-bound phase: masked mean over embs [64,2048,1024] f32 (~0.5 GB) ->
// nontemporal float4 streaming, split over (batch x S-chunk) blocks.
// Compute phase: [64,2048]x[2048,1024] GEMM via V_WMMA_F32_16X16X32_F16,
// one 16x16 tile per wave, weights pre-packed into B fragments with 16 contiguous K per lane half.
// ---------------------------------------------------------------------------

typedef __attribute__((ext_vector_type(16))) _Float16 v16h;
typedef __attribute__((ext_vector_type(8)))  _Float16 v8h;
typedef __attribute__((ext_vector_type(8)))  float    v8f;
typedef __attribute__((ext_vector_type(4)))  float    vf4;

#define BB 64
#define SS 2048
#define DD 1024
#define NCHUNK 8
#define CHUNK (SS / NCHUNK)   // 256

// --------------------------- kernel 1: find eos/pad indices ----------------
__global__ void k_find_idx(const int* __restrict__ ids,
                           int* __restrict__ idx1, int* __restrict__ idx2) {
    const int b = blockIdx.x, tid = threadIdx.x;
    int m1 = SS, m2 = SS;
    for (int s = tid; s < SS; s += 256) {
        const int id = ids[b * SS + s];
        if (id == 2 && s < m1) m1 = s;   // first EOS
        if (id == 1 && s < m2) m2 = s;   // first PAD
    }
    __shared__ int sh1[256], sh2[256];
    sh1[tid] = m1; sh2[tid] = m2;
    __syncthreads();
    for (int off = 128; off > 0; off >>= 1) {
        if (tid < off) {
            sh1[tid] = sh1[tid] < sh1[tid + off] ? sh1[tid] : sh1[tid + off];
            sh2[tid] = sh2[tid] < sh2[tid + off] ? sh2[tid] : sh2[tid + off];
        }
        __syncthreads();
    }
    if (tid == 0) { idx1[b] = sh1[0]; idx2[b] = sh2[0] - 1; }
}

// --------------------------- kernel 2: streaming partial sums --------------
// grid = (NCHUNK, B), block = 256; thread owns float4 column slice d = tid*4.
__global__ void __launch_bounds__(256) k_partial(
        const float* __restrict__ embs,
        const int* __restrict__ idx1, const int* __restrict__ idx2,
        float* __restrict__ part) {
    const int chunk = blockIdx.x, b = blockIdx.y, tid = threadIdx.x;
    const int i1 = idx1[b], i2 = idx2[b];
    const int c0 = chunk * CHUNK, c1 = c0 + CHUNK;
    // sent1 tokens: [1, i1); sent2 tokens: [i1+2, i2). Intersect with chunk ->
    // two branch-free loops instead of per-token masking.
    const int r1lo = (1     > c0) ? 1     : c0;
    const int r1hi = (i1    < c1) ? i1    : c1;
    const int r2lo = (i1+2  > c0) ? i1+2  : c0;
    const int r2hi = (i2    < c1) ? i2    : c1;

    vf4 a1 = {0.f, 0.f, 0.f, 0.f}, a2 = {0.f, 0.f, 0.f, 0.f};
    const vf4* base = (const vf4*)(embs + (size_t)b * SS * DD) + tid;
#pragma unroll 4
    for (int s = r1lo; s < r1hi; ++s)
        a1 += __builtin_nontemporal_load(base + (size_t)s * (DD / 4));
#pragma unroll 4
    for (int s = r2lo; s < r2hi; ++s)
        a2 += __builtin_nontemporal_load(base + (size_t)s * (DD / 4));

    vf4* p = (vf4*)part + ((size_t)(b * NCHUNK + chunk) * 2) * (DD / 4) + tid;
    p[0]      = a1;
    p[DD / 4] = a2;
}

// --------------------------- kernel 3: reduce + mean + f32->f16 ------------
__global__ void k_finalize(const float* __restrict__ part,
                           const int* __restrict__ idx1, const int* __restrict__ idx2,
                           _Float16* __restrict__ Xh) {
    const int b = blockIdx.x, tid = threadIdx.x;
    const float inv1 = 1.0f / (float)(idx1[b] - 1);
    const float inv2 = 1.0f / (float)(idx2[b] - idx1[b] - 2);
    vf4 s1 = {0.f,0.f,0.f,0.f}, s2 = {0.f,0.f,0.f,0.f};
    const vf4* p = (const vf4*)part + (size_t)b * NCHUNK * 2 * (DD / 4) + tid;
#pragma unroll
    for (int c = 0; c < NCHUNK; ++c) {
        s1 += p[(size_t)c * 2 * (DD / 4)];
        s2 += p[(size_t)c * 2 * (DD / 4) + (DD / 4)];
    }
    _Float16* x1 = Xh + (size_t)b * 2 * DD + tid * 4;  // concat row [sent1|sent2]
    _Float16* x2 = x1 + DD;
#pragma unroll
    for (int j = 0; j < 4; ++j) {
        x1[j] = (_Float16)(s1[j] * inv1);
        x2[j] = (_Float16)(s2[j] * inv2);
    }
}

// --------------------------- kernel 4: pack W into B-fragment layout -------
// B fragment (16-bit, 32x16): lane = N (mod 16), lane>=16 group holds K+16;
// each lane's 16 halves are consecutive K. grid=(64 nTiles, 64 kSteps), 1 wave.
__global__ void k_pack_w(const float* __restrict__ Wsrc, _Float16* __restrict__ Wp) {
    const int nt = blockIdx.x, ks = blockIdx.y, lane = threadIdx.x; // 0..31
    const int row0 = ks * 32 + (lane >> 4) * 16;  // K base for this lane group
    const int col  = nt * 16 + (lane & 15);       // N
    v16h v;
#pragma unroll
    for (int e = 0; e < 16; ++e)
        v[e] = (_Float16)Wsrc[(size_t)(row0 + e) * DD + col];
    *(v16h*)(Wp + ((size_t)(nt * 64 + ks) * 32 + lane) * 16) = v;
}

// --------------------------- kernel 5: WMMA GEMM + bias + tanh -------------
// 256 waves total (32 blocks x 8 waves); wave -> one 16x16 tile of Y[64,1024].
__global__ void __launch_bounds__(256) k_gemm(
        const _Float16* __restrict__ Xh, const _Float16* __restrict__ Wp,
        const float* __restrict__ bias, float* __restrict__ H) {
    const int tid  = threadIdx.x;
    const int wave = (blockIdx.x * 256 + tid) >> 5;   // 0..255
    const int lane = tid & 31;
    const int mT = wave >> 6;        // 0..3
    const int nT = wave & 63;        // 0..63
    const int lp = lane & 15, half = lane >> 4;

    const _Float16* arow = Xh + (size_t)(mT * 16 + lp) * 2048;
    const v16h* bptr = (const v16h*)Wp + (size_t)nT * 2048 + lane;

    v8f c = {};
#pragma unroll 4
    for (int ks = 0; ks < 64; ++ks) {
        const int k0 = ks * 32;
        // A fragment (16x32 f16): lanes 0-15 -> K {0..15}, 16-31 -> K {16..31}
        union { v16h v; v8h h[2]; } a;
        a.v = *(const v16h*)(arow + k0 + half * 16);
        const v16h bfrag = bptr[ks * 32];
        c = __builtin_amdgcn_wmma_f32_16x16x32_f16(
                /*neg_a=*/false, a.v, /*neg_b=*/false, bfrag,
                /*c_mod=*/(short)0, c, /*reuse_a=*/false, /*reuse_b=*/false);
    }
    // C/D layout: VGPR v holds M = half*8 + v, lane lp holds N.
    const int n    = nT * 16 + lp;
    const int mout = mT * 16 + half * 8;
    const float bn = bias[n];
#pragma unroll
    for (int v = 0; v < 8; ++v)
        H[(size_t)(mout + v) * DD + n] = tanhf(c[v] + bn);
}

// --------------------------- kernel 6: final 1024 -> 2 projection ----------
__global__ void k_logits(const float* __restrict__ H, const float* __restrict__ ow,
                         const float* __restrict__ ob, float* __restrict__ out) {
    const int b = blockIdx.x, tid = threadIdx.x;
    float p0 = 0.f, p1 = 0.f;
    for (int d = tid; d < DD; d += 256) {
        const float h = H[(size_t)b * DD + d];
        p0 += h * ow[d * 2 + 0];
        p1 += h * ow[d * 2 + 1];
    }
    __shared__ float s0[256], s1[256];
    s0[tid] = p0; s1[tid] = p1;
    __syncthreads();
    for (int off = 128; off > 0; off >>= 1) {
        if (tid < off) { s0[tid] += s0[tid + off]; s1[tid] += s1[tid + off]; }
        __syncthreads();
    }
    if (tid == 0) {
        out[b * 2 + 0] = s0[0] + ob[0];
        out[b * 2 + 1] = s1[0] + ob[1];
    }
}

// ---------------------------------------------------------------------------
extern "C" void kernel_launch(void* const* d_in, const int* in_sizes, int n_in,
                              void* d_out, int out_size, void* d_ws, size_t ws_size,
                              hipStream_t stream) {
    const float* embs    = (const float*)d_in[0];   // [64,2048,1024] f32
    const int*   ids     = (const int*)  d_in[1];   // [64,2048] i32
    const float* dense_w = (const float*)d_in[2];   // [2048,1024] f32
    const float* dense_b = (const float*)d_in[3];   // [1024] f32
    const float* out_w   = (const float*)d_in[4];   // [1024,2] f32
    const float* out_b   = (const float*)d_in[5];   // [2] f32
    float*       out     = (float*)d_out;           // [64,2] f32
    (void)in_sizes; (void)n_in; (void)out_size; (void)ws_size;

    // Workspace layout (256B-aligned offsets), total ~8.6 MB.
    char* ws = (char*)d_ws;
    int*      idx1 = (int*)(ws);                                   //   256 B
    int*      idx2 = (int*)(ws + 256);                             //   256 B
    float*    part = (float*)(ws + 512);                           //  4 MiB [B][8][2][D]
    _Float16* Xh   = (_Float16*)(ws + 512 + 4194304);              // 256 KiB [64,2048] f16
    _Float16* Wp   = (_Float16*)(ws + 512 + 4194304 + 262144);     //  4 MiB packed B frags
    float*    Hbuf = (float*)(ws + 512 + 4194304 + 262144 + 4194304); // 256 KiB [64,1024]

    k_find_idx<<<BB, 256, 0, stream>>>(ids, idx1, idx2);
    k_partial <<<dim3(NCHUNK, BB), 256, 0, stream>>>(embs, idx1, idx2, part);
    k_finalize<<<BB, 256, 0, stream>>>(part, idx1, idx2, Xh);
    k_pack_w  <<<dim3(64, 64), 32, 0, stream>>>(dense_w, Wp);
    k_gemm    <<<32, 256, 0, stream>>>(Xh, Wp, dense_b, Hbuf);
    k_logits  <<<BB, 256, 0, stream>>>(Hbuf, out_w, out_b, out);
}
